// HM_LSTM_84078279787177
// MI455X (gfx1250) — matrix operand order repaired and checked by reading, not compiled, not run on hardware
//
#include <hip/hip_runtime.h>
#include <hip/hip_bf16.h>

// ---------------------------------------------------------------------------
// HM-LSTM on gfx1250: persistent-grid recurrent kernel with bf16 WMMA GEMMs.
//   B=64, T=2048, INPUT=256, H=512, G=4H+1=2049 (padded to 2064 = 129 Mtiles)
//
// Design (reasoned from MI455X specs, compile-only):
//  * 2.06 TFLOP total over a hard serial recurrence of T=2048 steps -> latency
//    bound, not HBM bound (HBM traffic ~1 GB ~= 43 us at 23.3 TB/s).
//  * bf16 WMMA (K=32/instr, fp32 accumulate) vs fp32 WMMA (K=4/instr): 8x
//    fewer matrix instructions on the critical path.
//  * Weights packed ONCE into a contiguous-k (k = 16h+s) 16-bit A-operand lane layout
//    (16x32 tiles); h state kept col-major bf16 so it IS the B-operand layout.
//    Both stream from L2 (weights 15.9 MB << 192 MB L2).
//  * Per-batch-column gate scales z / z_b applied to WMMA partial products
//    (each lane owns one output column) instead of rescaling B every step.
//  * x_t pre-staged to bf16 B-layout in a 2x32KB double buffer during the
//    previous step's elementwise phase -> no fp32 cvt in the GEMM hot loop.
//  * Persistent grid (64 WG x 256 thr = 512 waves; 516 output tiles/GEMM),
//    software grid barrier between GEMM and elementwise phases.
// ---------------------------------------------------------------------------

typedef __attribute__((ext_vector_type(16))) __bf16 v16bf;
typedef __attribute__((ext_vector_type(8)))  float  v8f;
typedef __bf16 bf16_t;

#define HM_B      64
#define HM_T      2048
#define HM_IN     256
#define HM_H      512
#define HM_G      2049
#define HM_GPAD   2064
#define HM_MT     129          // 2064/16
#define NWG       64
#define TPB       256

// ---------------------- parameter bundle -----------------------------------
struct KParams {
    const float*  x;           // (64,2048,256)
    const bf16_t* A[3];        // packed A-operand weights per cell
    const float*  bias[3];     // (G,)
    float*  Hf[3];             // fp32 h state, col-major (b*512 + j)
    bf16_t* Hb[3];             // bf16 h state, col-major (b*512 + j) == B operand
    float*  C[3];              // fp32 c state
    float*  Z[3];              // 128 floats each: two 64-wide halves (parity)
    bf16_t* xs;                // staged x_t, 2 x (64*256) bf16, B-operand layout
    float*  S;                 // (2064 x 64) fp32 gate pre-activations
    unsigned* cnt;
    unsigned* gen;
    float* oH[3];
    float* oZ1; float* oZ2; float* oZ3;
    float* oHid;
};

// ---------------------- grid-wide barrier -----------------------------------
__device__ __forceinline__ void gsync(unsigned* cnt, unsigned* gen, unsigned nwg) {
    __syncthreads();
    if (threadIdx.x == 0) {
        __threadfence();
        unsigned g = __atomic_load_n(gen, __ATOMIC_RELAXED);
        unsigned r = atomicAdd(cnt, 1u);
        if (r == nwg - 1u) {
            __atomic_store_n(cnt, 0u, __ATOMIC_RELAXED);
            __threadfence();
            atomicAdd(gen, 1u);
        } else {
            while (__atomic_load_n(gen, __ATOMIC_RELAXED) == g) {
                __builtin_amdgcn_s_sleep(2);
            }
        }
        __threadfence();
    }
    __syncthreads();
}

// ---------------------- WMMA helper -----------------------------------------
__device__ __forceinline__ v8f wmma_bf16(v16bf a, v16bf b, v8f c) {
    return __builtin_amdgcn_wmma_f32_16x16x32_bf16(
        false, a, false, b, (short)0, c, false, false);
}

// ---------------------- x staging (fp32 -> bf16 B-operand layout) ------------
__device__ __forceinline__ void stage_x(const KParams& p, int t) {
    const int gtid = blockIdx.x * blockDim.x + threadIdx.x;
    const int tot  = gridDim.x * blockDim.x;
    bf16_t* __restrict__ dst = p.xs + (size_t)(t & 1) * (HM_B * HM_IN);
    const float* __restrict__ src = p.x;
    for (int e = gtid; e < HM_B * HM_IN; e += tot) {
        const int b = e >> 8, k = e & 255;        // dst[b*256+k]
        dst[e] = (bf16_t)src[((size_t)b * HM_T + t) * HM_IN + k];
    }
}

// ---------------------- GEMM phase (per cell) --------------------------------
// Per-cell K sections (in 32-wide K-tiles):
//   cell0: rec=h1(16) | bu=x staged(8)  | td=h2(16)   KT=40
//   cell1: rec=h2(16) | bu=h1_new(16)   | td=h3(16)   KT=48
//   cell2: rec=h3(16) | bu=h2_new(16)                 KT=32
template <int CELL>
__device__ __forceinline__ void gemm_phase(const KParams& p, int t,
                                           int half_prev, int half_new) {
    const int lane = threadIdx.x & 31;
    const int gw   = blockIdx.x * (blockDim.x >> 5) + (threadIdx.x >> 5);
    const int nw   = gridDim.x * (blockDim.x >> 5);

    constexpr int KT     = (CELL == 0) ? 40 : (CELL == 1) ? 48 : 32;
    constexpr int KT_BU  = (CELL == 0) ? 8 : 16;
    constexpr bool HAS_TD = (CELL != 2);

    const bf16_t* __restrict__ Apack = p.A[CELL];
    const bf16_t* __restrict__ hrec  = p.Hb[(CELL == 0) ? 0 : (CELL == 1) ? 1 : 2];
    const bf16_t* __restrict__ hbu   =
        (CELL == 0) ? p.xs + (size_t)(t & 1) * (HM_B * HM_IN)
      : (CELL == 1) ? p.Hb[0] : p.Hb[1];
    const bf16_t* __restrict__ htd   =
        (CELL == 0) ? p.Hb[1] : (CELL == 1) ? p.Hb[2] : nullptr;
    const float*  __restrict__ bias  = p.bias[CELL];
    const float*  __restrict__ zbArr = (CELL == 1) ? p.Z[0] + half_new * 64
                                     : (CELL == 2) ? p.Z[1] + half_new * 64 : nullptr;
    const float*  __restrict__ ztArr = (CELL == 0) ? p.Z[0] + half_prev * 64
                                     : (CELL == 1) ? p.Z[1] + half_prev * 64 : nullptr;

    constexpr int BU_STRIDE = (CELL == 0) ? HM_IN : HM_H;
    const int kHalf = (lane < 16) ? 0 : 16;

    for (int tile = gw; tile < HM_MT * 4; tile += nw) {
        const int mt  = tile >> 2;
        const int nt  = tile & 3;
        const int col = nt * 16 + (lane & 15);
        const bf16_t* __restrict__ Abase =
            Apack + (size_t)mt * KT * 512 + (size_t)lane * 16;

        v8f accR = {}; v8f accB = {}; v8f accT = {};

        // --- recurrent section (K = 512)
#pragma unroll
        for (int s = 0; s < 16; ++s) {
            v16bf a = *(const v16bf*)(Abase + (size_t)s * 512);
            v16bf b = *(const v16bf*)(hrec + (size_t)col * HM_H + s * 32 + kHalf);
            accR = wmma_bf16(a, b, accR);
        }
        // --- bottom-up section (x staged for cell0, h below otherwise)
#pragma unroll
        for (int s = 0; s < KT_BU; ++s) {
            v16bf a = *(const v16bf*)(Abase + (size_t)(16 + s) * 512);
            v16bf b = *(const v16bf*)(hbu + (size_t)col * BU_STRIDE + s * 32 + kHalf);
            accB = wmma_bf16(a, b, accB);
        }
        // --- top-down section
        if constexpr (HAS_TD) {
#pragma unroll
            for (int s = 0; s < 16; ++s) {
                v16bf a = *(const v16bf*)(Abase + (size_t)(16 + KT_BU + s) * 512);
                v16bf b = *(const v16bf*)(htd + (size_t)col * HM_H + s * 32 + kHalf);
                accT = wmma_bf16(a, b, accT);
            }
        }

        const float zb = (CELL == 0) ? 1.0f : zbArr[col];
        const float zt = HAS_TD ? ztArr[col] : 0.0f;
        const int rbase = mt * 16 + ((lane < 16) ? 0 : 8);
#pragma unroll
        for (int r = 0; r < 8; ++r) {
            const int row = rbase + r;
            const float bv = (row < HM_G) ? bias[row] : 0.0f;
            float sv = accR[r] + zb * accB[r] + bv;
            if constexpr (HAS_TD) sv += zt * accT[r];
            p.S[(size_t)row * 64 + col] = sv;
        }
    }
}

// ---------------------- elementwise phase (per cell) -------------------------
template <int CELL>
__device__ __forceinline__ void ew_phase(const KParams& p, int t,
                                         int half_prev, int half_new) {
    const int gtid = blockIdx.x * blockDim.x + threadIdx.x;
    const int tot  = gridDim.x * blockDim.x;

    const float* __restrict__ zP    = p.Z[CELL] + half_prev * 64;
    float*       __restrict__ zN    = p.Z[CELL] + half_new * 64;
    const float* __restrict__ zbArr = (CELL == 1) ? p.Z[0] + half_new * 64
                                    : (CELL == 2) ? p.Z[1] + half_new * 64 : nullptr;
    float*  __restrict__ Hf  = p.Hf[CELL];
    bf16_t* __restrict__ Hbv = p.Hb[CELL];
    float*  __restrict__ C   = p.C[CELL];
    float*  __restrict__ oH  = p.oH[CELL];
    const float* __restrict__ S = p.S;

    for (int e = gtid; e < HM_H * HM_B; e += tot) {
        const int b = e & 63, j = e >> 6;
        const float sf = S[(size_t)(0 * HM_H + j) * 64 + b];
        const float si = S[(size_t)(1 * HM_H + j) * 64 + b];
        const float so = S[(size_t)(2 * HM_H + j) * 64 + b];
        const float sg = S[(size_t)(3 * HM_H + j) * 64 + b];
        const float z  = zP[b];
        const float zb = (CELL == 0) ? 1.0f : zbArr[b];

        const float f = 1.0f / (1.0f + __expf(-sf));
        const float i = 1.0f / (1.0f + __expf(-si));
        const float o = 1.0f / (1.0f + __expf(-so));
        const float g = tanhf(sg);

        const float c  = C[(size_t)b * HM_H + j];
        const float ig = i * g;
        const float cn = z * ig + (1.0f - z) * ((1.0f - zb) * c + zb * (f * c + ig));
        const float th = o * tanhf(cn);
        const float hp = Hf[(size_t)b * HM_H + j];
        const float hn = z * th + (1.0f - z) * ((1.0f - zb) * hp + zb * th);

        C[(size_t)b * HM_H + j]  = cn;
        Hf[(size_t)b * HM_H + j] = hn;
        Hbv[(size_t)b * HM_H + j] = (bf16_t)hn;
        __builtin_nontemporal_store(
            hn, oH + (size_t)b * (HM_T * HM_H) + (size_t)t * HM_H + j);
    }
    if (gtid < HM_B) {
        const int b = gtid;
        const float sz = S[(size_t)(4 * HM_H) * 64 + b];
        float zh = (sz + 1.0f) * 0.5f;              // hard_sigm, A=1
        zh = fminf(fmaxf(zh, 0.0f), 1.0f);
        const float zn = (zh > 0.5f) ? 1.0f : 0.0f; // st_bound forward value
        zN[b] = zn;
        if (CELL == 0) p.oZ1[(size_t)b * HM_T + t] = zn;
        if (CELL == 1) {
            p.oZ2[(size_t)b * HM_T + t] = zn;
            p.oZ3[(size_t)b * HM_T + t] = zn;
        }
    }
}

// ---------------------- persistent recurrent kernel --------------------------
__global__ void __launch_bounds__(TPB)
hmlstm_persistent(KParams p) {
    const unsigned nwg = gridDim.x;
    stage_x(p, 0);
    gsync(p.cnt, p.gen, nwg);
    for (int t = 0; t < HM_T; ++t) {
        const int hp = t & 1, hn = (t + 1) & 1;
        gemm_phase<0>(p, t, hp, hn); gsync(p.cnt, p.gen, nwg);
        ew_phase<0>(p, t, hp, hn);   gsync(p.cnt, p.gen, nwg);
        gemm_phase<1>(p, t, hp, hn); gsync(p.cnt, p.gen, nwg);
        ew_phase<1>(p, t, hp, hn);   gsync(p.cnt, p.gen, nwg);
        gemm_phase<2>(p, t, hp, hn); gsync(p.cnt, p.gen, nwg);
        ew_phase<2>(p, t, hp, hn);
        if (t + 1 < HM_T) stage_x(p, t + 1);   // prep next step's x (other parity)
        gsync(p.cnt, p.gen, nwg);
    }
    // final hidden tuple: (h1,c1,z1,h2,c2,z2,h3,c3,z3), each (dim, B)
    const int gtid = blockIdx.x * blockDim.x + threadIdx.x;
    const int tot  = gridDim.x * blockDim.x;
    for (int L = 0; L < 3; ++L) {
        float* base = p.oHid + (size_t)L * (2 * HM_H * HM_B + HM_B);
        const float* Hf = p.Hf[L];
        const float* C  = p.C[L];
        for (int e = gtid; e < HM_H * HM_B; e += tot) {
            const int j = e >> 6, b = e & 63;
            base[(size_t)j * 64 + b]               = Hf[(size_t)b * HM_H + j];
            base[HM_H * HM_B + (size_t)j * 64 + b] = C[(size_t)b * HM_H + j];
        }
        if (gtid < HM_B) base[2 * HM_H * HM_B + gtid] = p.Z[L][gtid]; // half 0
    }
}

// ---------------------- weight packing (fp32 -> bf16, WMMA A layout) ---------
// 16-bit A 16x32 tile: lane<16 -> row=lane, K = j (0..15);
//                      lane>=16 -> row=lane-16, K = 16 + j.
__global__ void pack_weights(const float* __restrict__ src, bf16_t* __restrict__ dst,
                             int Grows, int Ksrc, int KTsrc, int KTtot, int ktOff) {
    const int tile = blockIdx.x;            // over MT * KTsrc
    const int mt = tile / KTsrc;
    const int kt = tile % KTsrc;
    const size_t base = ((size_t)mt * KTtot + ktOff + kt) * 512;
    for (int i = threadIdx.x; i < 512; i += blockDim.x) {
        const int lane = i >> 4, j = i & 15, v = j >> 1, hb = j & 1;
        int kk, rl;
        if (lane < 16) { rl = lane;      kk = 2 * v + hb; }
        else           { rl = lane - 16; kk = 16 + 2 * v + hb; }
        const int row = mt * 16 + rl;
        const int colc = kt * 32 + kk;
        const float val = (row < Grows && colc < Ksrc)
                              ? src[(size_t)row * Ksrc + colc] : 0.0f;
        dst[base + i] = (bf16_t)val;
    }
}

// ---------------------- state / barrier init ---------------------------------
__global__ void init_state(float* state, int nfloats, unsigned* cnt, unsigned* gen) {
    const int i = blockIdx.x * blockDim.x + threadIdx.x;
    if (i == 0) { *cnt = 0u; *gen = 0u; }
    for (int e = i; e < nfloats; e += gridDim.x * blockDim.x) state[e] = 0.0f;
}

// ---------------------- host launcher ----------------------------------------
extern "C" void kernel_launch(void* const* d_in, const int* in_sizes, int n_in,
                              void* d_out, int out_size, void* d_ws, size_t ws_size,
                              hipStream_t stream) {
    (void)in_sizes; (void)n_in; (void)out_size; (void)ws_size;
    char* ws = (char*)d_ws;
    auto align256 = [](size_t x) { return (x + 255) & ~(size_t)255; };

    size_t off = 0;
    unsigned* cnt = (unsigned*)(ws + off);
    unsigned* gen = cnt + 1;
    off = 256;

    const size_t A0e = (size_t)HM_MT * 40 * 512;
    const size_t A1e = (size_t)HM_MT * 48 * 512;
    const size_t A2e = (size_t)HM_MT * 32 * 512;
    bf16_t* A0 = (bf16_t*)(ws + off); off = align256(off + A0e * 2);
    bf16_t* A1 = (bf16_t*)(ws + off); off = align256(off + A1e * 2);
    bf16_t* A2 = (bf16_t*)(ws + off); off = align256(off + A2e * 2);

    const size_t stateOff = off;
    float* Hf[3]; float* Cc[3]; bf16_t* Hb[3]; float* Zz[3];
    for (int L = 0; L < 3; ++L) { Hf[L] = (float*)(ws + off); off += (size_t)HM_H * HM_B * 4; }
    for (int L = 0; L < 3; ++L) { Cc[L] = (float*)(ws + off); off += (size_t)HM_H * HM_B * 4; }
    for (int L = 0; L < 3; ++L) { Hb[L] = (bf16_t*)(ws + off); off += (size_t)HM_H * HM_B * 2; }
    for (int L = 0; L < 3; ++L) { Zz[L] = (float*)(ws + off); off += 128 * 4; }
    const size_t stateFloats = (off - stateOff) / 4;
    off = align256(off);
    float* Sbuf = (float*)(ws + off); off += (size_t)HM_GPAD * 64 * 4;
    off = align256(off);
    bf16_t* Xs = (bf16_t*)(ws + off); off += (size_t)2 * HM_B * HM_IN * 2;

    // ---- init barrier + state
    init_state<<<256, 256, 0, stream>>>((float*)(ws + stateOff), (int)stateFloats, cnt, gen);

    // ---- pack weights (d_in order: inputs,U11_1,U21_1,W01_1,b_1,
    //                    U11_2,U21_2,W01_2,b_2, U11_3,W01_3,b_3)
    const float* U11_1 = (const float*)d_in[1];
    const float* U21_1 = (const float*)d_in[2];
    const float* W01_1 = (const float*)d_in[3];
    const float* U11_2 = (const float*)d_in[5];
    const float* U21_2 = (const float*)d_in[6];
    const float* W01_2 = (const float*)d_in[7];
    const float* U11_3 = (const float*)d_in[9];
    const float* W01_3 = (const float*)d_in[10];

    pack_weights<<<HM_MT * 16, 256, 0, stream>>>(U11_1, A0, HM_G, 512, 16, 40, 0);
    pack_weights<<<HM_MT *  8, 256, 0, stream>>>(W01_1, A0, HM_G, 256,  8, 40, 16);
    pack_weights<<<HM_MT * 16, 256, 0, stream>>>(U21_1, A0, HM_G, 512, 16, 40, 24);
    pack_weights<<<HM_MT * 16, 256, 0, stream>>>(U11_2, A1, HM_G, 512, 16, 48, 0);
    pack_weights<<<HM_MT * 16, 256, 0, stream>>>(W01_2, A1, HM_G, 512, 16, 48, 16);
    pack_weights<<<HM_MT * 16, 256, 0, stream>>>(U21_2, A1, HM_G, 512, 16, 48, 32);
    pack_weights<<<HM_MT * 16, 256, 0, stream>>>(U11_3, A2, HM_G, 512, 16, 32, 0);
    pack_weights<<<HM_MT * 16, 256, 0, stream>>>(W01_3, A2, HM_G, 512, 16, 32, 16);

    // ---- params
    KParams kp;
    kp.x = (const float*)d_in[0];
    kp.A[0] = A0; kp.A[1] = A1; kp.A[2] = A2;
    kp.bias[0] = (const float*)d_in[4];
    kp.bias[1] = (const float*)d_in[8];
    kp.bias[2] = (const float*)d_in[11];
    for (int L = 0; L < 3; ++L) {
        kp.Hf[L] = Hf[L]; kp.Hb[L] = Hb[L]; kp.C[L] = Cc[L]; kp.Z[L] = Zz[L];
    }
    kp.xs = Xs; kp.S = Sbuf; kp.cnt = cnt; kp.gen = gen;

    float* out = (float*)d_out;
    const size_t SH = (size_t)HM_B * HM_T * HM_H;
    const size_t SZ = (size_t)HM_B * HM_T;
    kp.oH[0] = out;
    kp.oH[1] = out + SH;
    kp.oH[2] = out + 2 * SH;
    kp.oZ1 = out + 3 * SH;
    kp.oZ2 = out + 3 * SH + SZ;
    kp.oZ3 = out + 3 * SH + 2 * SZ;
    kp.oHid = out + 3 * SH + 3 * SZ;

    hmlstm_persistent<<<NWG, TPB, 0, stream>>>(kp);
}
